// DTDRLinear_88441966559314
// MI455X (gfx1250) — matrix operand order repaired and checked by reading, not compiled, not run on hardware
//
#include <hip/hip_runtime.h>

// Problem sizes (fixed by the reference)
#define TOKENS 8192
#define IN_F   4096
#define OUT_F  4096

// GEMM tiling
#define BM 128      // block tile M (tokens)
#define BN 64       // block tile N (outputs)
#define KT 64       // K depth staged per TDM load

typedef __attribute__((ext_vector_type(16))) _Float16 v16h;
typedef __attribute__((ext_vector_type(8)))  _Float16 v8h;
typedef __attribute__((ext_vector_type(8)))  float    v8f;
typedef __attribute__((ext_vector_type(4)))  unsigned int v4u;
typedef __attribute__((ext_vector_type(8)))  int      v8i;
typedef __attribute__((ext_vector_type(4)))  int      v4i;

// ---------------------------------------------------------------------------
// Kernel 1: orthonormal FWHT along rows of x (f32 in LDS), emit f16.
// ---------------------------------------------------------------------------
__global__ __launch_bounds__(256) void fwht_kernel(const float* __restrict__ x,
                                                   _Float16* __restrict__ xh) {
  __shared__ float smem[IN_F];
  const int row = blockIdx.x;
  const float* xr = x + (size_t)row * IN_F;

  for (int i = threadIdx.x; i < IN_F; i += 256) smem[i] = xr[i];
  __syncthreads();

  for (int h = 1; h < IN_F; h <<= 1) {
    for (int p = threadIdx.x; p < IN_F / 2; p += 256) {
      const int i = ((p & ~(h - 1)) << 1) | (p & (h - 1));
      const int j = i + h;
      const float a = smem[i];
      const float b = smem[j];
      smem[i] = a + b;
      smem[j] = a - b;
    }
    __syncthreads();
  }

  _Float16* out = xh + (size_t)row * IN_F;
  const float inv = 0.015625f;  // 1/sqrt(4096)
  for (int i = threadIdx.x; i < IN_F; i += 256)
    out[i] = (_Float16)(smem[i] * inv);
}

// ---------------------------------------------------------------------------
// Kernel 2: Q_tilde (int32 in [-127,127]) -> f16 (exact).
// ---------------------------------------------------------------------------
__global__ __launch_bounds__(256) void cvtq_kernel(const int* __restrict__ q,
                                                   _Float16* __restrict__ qh) {
  const size_t i = (size_t)blockIdx.x * 256u + threadIdx.x;
  qh[i] = (_Float16)(float)q[i];
}

// ---------------------------------------------------------------------------
// TDM: issue a 2-D tile load (Tensor Data Mover) global -> LDS.
//   tile = tile1 rows x tile0 elements, element size 2 bytes (f16),
//   row stride in the source tensor = stride0 elements.
// Descriptor packing per CDNA5 ISA (D# groups 0..3).
// amdgpu-toolchain (clang-23) builtin form: 6 args
//   (v4u g0, v8i g1, v4i g2, v4i g3, v8i extra, i32 cpol)
// ---------------------------------------------------------------------------
__device__ __forceinline__ void tdm_load_2d_f16(unsigned int lds_addr,
                                                unsigned long long gaddr,
                                                unsigned int tdim0, unsigned int tdim1,
                                                unsigned int tile0, unsigned int tile1,
                                                unsigned long long stride0) {
  v4u g0;
  g0.x = 1u;                                        // count=1 (valid), user mode
  g0.y = lds_addr;                                  // lds_addr [63:32]
  g0.z = (unsigned int)gaddr;                       // global_addr [95:64]
  g0.w = (unsigned int)(gaddr >> 32) | (2u << 30);  // global_addr[56:32] | type=2

  v8i g1;
  g1[0] = (int)(1u << 16);                                  // data_size=1 (2B); mask=0
  g1[1] = (int)((tdim0 & 0xFFFFu) << 16);                   // tensor_dim0[15:0]
  g1[2] = (int)(((tdim0 >> 16) & 0xFFFFu) |
                ((tdim1 & 0xFFFFu) << 16));                 // tensor_dim0[31:16] | tensor_dim1[15:0]
  g1[3] = (int)(((tdim1 >> 16) & 0xFFFFu) |
                ((tile0 & 0xFFFFu) << 16));                 // tensor_dim1[31:16] | tile_dim0
  g1[4] = (int)(tile1 & 0xFFFFu);                           // tile_dim1 (tile_dim2=0)
  g1[5] = (int)(unsigned int)(stride0 & 0xFFFFFFFFu);       // tensor_dim0_stride[31:0]
  g1[6] = (int)(unsigned int)((stride0 >> 32) & 0xFFFFu);   // stride[47:32] (dim1_stride=0)
  g1[7] = 0;

  v4i g2 = {0, 0, 0, 0};  // unused (2-D tensor)
  v4i g3 = {0, 0, 0, 0};
  v8i g4 = {0, 0, 0, 0, 0, 0, 0, 0};  // extra descriptor words (unused, 2-D)

  __builtin_amdgcn_tensor_load_to_lds(g0, g1, g2, g3, g4, 0);
}

// ---------------------------------------------------------------------------
// Kernel 3: y = xh @ Qf16^T (f32 accum), epilogue *s[n] + bias[n].
// 256 threads = 8 wave32 waves (4 along M x 2 along N); block tile 128x64.
// A/B tiles staged to LDS by TDM (double-buffered, TENSORcnt-synced);
// each wave runs a 32x32 tile = 2x2 V_WMMA_F32_16X16X32_F16 per K-step.
// LDS: A 2*128*64*2B = 32 KB, B 2*64*64*2B = 16 KB -> 48 KB.
// ---------------------------------------------------------------------------
__global__ __launch_bounds__(256) void gemm_kernel(const _Float16* __restrict__ A,
                                                   const _Float16* __restrict__ B,
                                                   const float* __restrict__ s,
                                                   const float* __restrict__ bias,
                                                   float* __restrict__ y) {
  __shared__ _Float16 shA[2][BM * KT];
  __shared__ _Float16 shB[2][BN * KT];

  const int lane = threadIdx.x & 31;
  const int wave = threadIdx.x >> 5;
  const int wm = wave & 3;   // 4 waves along M
  const int wn = wave >> 2;  // 2 waves along N
  const int mBlock = blockIdx.y * BM;
  const int nBlock = blockIdx.x * BN;
  const int l16 = lane & 15;
  const int sel = lane >> 4;

  // Base LDS fragment addresses (row = l16; k = 16*sel + s, contiguous, same for A and B), row stride = KT.
  const int aRow = wm * 32 + l16;           // + 16 for the second M tile
  const int bRow = wn * 32 + l16;           // + 16 for the second N tile
  const int aOff = aRow * KT + sel * 16;    // A: one 16-half chunk (two 8-half loads)
  const int bOff = bRow * KT + sel * 16;    // B: one 16-half chunk

  const unsigned long long gA =
      (unsigned long long)(uintptr_t)(A + (size_t)mBlock * IN_F);
  const unsigned long long gB =
      (unsigned long long)(uintptr_t)(B + (size_t)nBlock * IN_F);

  const unsigned int ldsA0 = (unsigned int)(uintptr_t)&shA[0][0];
  const unsigned int ldsA1 = (unsigned int)(uintptr_t)&shA[1][0];
  const unsigned int ldsB0 = (unsigned int)(uintptr_t)&shB[0][0];
  const unsigned int ldsB1 = (unsigned int)(uintptr_t)&shB[1][0];

  v8f acc[2][2] = {};

  const int NKB = IN_F / KT;  // 64 K-tiles

  // Prologue: wave 0 issues the first A/B tile pair into buffer 0.
  if (wave == 0) {
    tdm_load_2d_f16(ldsA0, gA, IN_F, TOKENS, KT, BM, IN_F);
    tdm_load_2d_f16(ldsB0, gB, IN_F, OUT_F, KT, BN, IN_F);
  }

  for (int it = 0; it < NKB; ++it) {
    const int cur = it & 1;
    if (wave == 0) {
      if (it + 1 < NKB) {
        const unsigned long long kByte = (unsigned long long)(it + 1) * KT * 2u;
        tdm_load_2d_f16(cur ? ldsA0 : ldsA1, gA + kByte, IN_F, TOKENS, KT, BM, IN_F);
        tdm_load_2d_f16(cur ? ldsB0 : ldsB1, gB + kByte, IN_F, OUT_F, KT, BN, IN_F);
        __builtin_amdgcn_s_wait_tensorcnt(2);  // current pair (issued last iter) done
      } else {
        __builtin_amdgcn_s_wait_tensorcnt(0);
      }
    }
    __syncthreads();  // LDS tile visible to all waves

    const _Float16* sA = &shA[cur][aOff];
    const _Float16* sB = &shB[cur][bOff];

    #pragma unroll
    for (int ks = 0; ks < KT; ks += 32) {
      const v8h a0lo = *(const v8h*)(sA + ks);
      const v8h a0hi = *(const v8h*)(sA + ks + 8);
      const v8h a1lo = *(const v8h*)(sA + 16 * KT + ks);
      const v8h a1hi = *(const v8h*)(sA + 16 * KT + ks + 8);
      const v16h a0 = __builtin_shufflevector(a0lo, a0hi, 0,1,2,3,4,5,6,7,8,9,10,11,12,13,14,15);
      const v16h a1 = __builtin_shufflevector(a1lo, a1hi, 0,1,2,3,4,5,6,7,8,9,10,11,12,13,14,15);

      const v16h b0 = *(const v16h*)(sB + ks);
      const v16h b1 = *(const v16h*)(sB + 16 * KT + ks);

      acc[0][0] = __builtin_amdgcn_wmma_f32_16x16x32_f16(false, a0, false, b0,
                                                         (short)0, acc[0][0], false, false);
      acc[0][1] = __builtin_amdgcn_wmma_f32_16x16x32_f16(false, a0, false, b1,
                                                         (short)0, acc[0][1], false, false);
      acc[1][0] = __builtin_amdgcn_wmma_f32_16x16x32_f16(false, a1, false, b0,
                                                         (short)0, acc[1][0], false, false);
      acc[1][1] = __builtin_amdgcn_wmma_f32_16x16x32_f16(false, a1, false, b1,
                                                         (short)0, acc[1][1], false, false);
    }
    __syncthreads();  // done reading `cur` before TDM overwrites it next+1 iter
  }

  // Epilogue. C/D layout: lane holds column n = l16; VGPR r holds row
  // m = sel*8 + r of the 16x16 tile.
  #pragma unroll
  for (int i = 0; i < 2; ++i) {
    #pragma unroll
    for (int j = 0; j < 2; ++j) {
      const int col = nBlock + wn * 32 + j * 16 + l16;
      const float sc = s[col];
      const float bi = bias[col];
      const int mTop = mBlock + wm * 32 + i * 16 + sel * 8;
      float* yp = y + (size_t)mTop * OUT_F + col;
      #pragma unroll
      for (int r = 0; r < 8; ++r)
        yp[(size_t)r * OUT_F] = acc[i][j][r] * sc + bi;
    }
  }
}

// ---------------------------------------------------------------------------
// Launcher. Inputs: x f32, Q_tilde i32, s_tilde f32(4096), bias f32(4096).
// Workspace: [xh f16: 64 MiB][qh f16: 32 MiB].
// ---------------------------------------------------------------------------
extern "C" void kernel_launch(void* const* d_in, const int* in_sizes, int n_in,
                              void* d_out, int out_size, void* d_ws, size_t ws_size,
                              hipStream_t stream) {
  const float* x    = (const float*)d_in[0];
  const int*   q    = (const int*)d_in[1];
  const float* s    = (const float*)d_in[2];
  const float* bias = (const float*)d_in[3];
  float*       y    = (float*)d_out;

  _Float16* xh = (_Float16*)d_ws;
  _Float16* qh = (_Float16*)((char*)d_ws + (size_t)TOKENS * IN_F * sizeof(_Float16));

  fwht_kernel<<<TOKENS, 256, 0, stream>>>(x, xh);
  cvtq_kernel<<<(OUT_F * IN_F) / 256, 256, 0, stream>>>(q, qh);

  dim3 grid(OUT_F / BN, TOKENS / BM);
  gemm_kernel<<<grid, 256, 0, stream>>>(xh, qh, s, bias, y);
}
